// Quantized8bitLinear_70866960384114
// MI455X (gfx1250) — matrix operand order repaired and checked by reading, not compiled, not run on hardware
//
#include <hip/hip_runtime.h>

// ---------------------------------------------------------------------------
// Quantized8bitLinear for MI455X (gfx1250, wave32, WMMA)
//   y[m,n] = sum_k x[m,k] * e4m3(W[n,k]) + e4m3(b[n])
// M=8192 (B*S), N=4096, K=4096.  Compute-bound f16 tensor GEMM.
// Data path: async global->LDS DMA (ASYNCcnt) double-buffered against the
// v_wmma_f32_16x16x32_f16 pack, one workgroup barrier per K-step.
// ---------------------------------------------------------------------------

typedef _Float16 v8h  __attribute__((ext_vector_type(8)));
typedef _Float16 v16h __attribute__((ext_vector_type(16)));
typedef float    v8f  __attribute__((ext_vector_type(8)));
typedef int      i4   __attribute__((ext_vector_type(4)));

constexpr int Mdim = 8192;   // B * S
constexpr int Ndim = 4096;   // D_OUT
constexpr int Kdim = 4096;   // D_IN

constexpr int BM = 128;      // block tile M (2 waves x 64)
constexpr int BN = 256;      // block tile N (4 waves x 64)
constexpr int BK = 32;       // K step (one f16 WMMA depth)
constexpr int LDA = BK + 8;  // padded LDS row stride (halves) -> 80B, conflict-free b128
constexpr int LDB = BK + 8;

#if defined(__has_builtin)
#if __has_builtin(__builtin_amdgcn_global_load_async_to_lds_b128) && \
    __has_builtin(__builtin_amdgcn_s_wait_asynccnt)
#define USE_ASYNC_LDS 1
#endif
#endif
#ifndef USE_ASYNC_LDS
#define USE_ASYNC_LDS 0
#endif

// builtin signature: (int4 as1*, int4 as3*, imm offset, imm cpol)
#define AS1_I4(p) ((__attribute__((address_space(1))) i4*)(p))
#define AS3_I4(p) ((__attribute__((address_space(3))) i4*)(p))

// --- fp8 e4m3fn round-trip emulation (RNE), evaluated in fp32 ---------------
__device__ __forceinline__ float quant_e4m3(float x) {
  float ax = fabsf(x);
  float y;
  if (ax < 0.015625f) {                 // below 2^-6: e4m3 subnormal, quantum 2^-9
    y = rintf(ax * 512.0f) * 0.001953125f;   // RNE via v_rndne
  } else {                              // normal: keep 3 mantissa bits, RNE w/ carry
    unsigned u   = __float_as_uint(ax);
    unsigned lsb = (u >> 20) & 1u;
    u += 0x7FFFFu + lsb;                // round-to-nearest-even into bit 20
    u &= 0xFFF00000u;
    y = __uint_as_float(u);
    y = fminf(y, 448.0f);               // e4m3fn finite max
  }
  return copysignf(y, x);
}

// --- W: fp32 -> e4m3 round -> f16 (exact), 8 elems/thread -------------------
__global__ __launch_bounds__(256) void quant_w_kernel(const float* __restrict__ w,
                                                      _Float16* __restrict__ w16) {
  size_t i = ((size_t)blockIdx.x * blockDim.x + threadIdx.x) * 8;
  float4 a = *reinterpret_cast<const float4*>(w + i);
  float4 b = *reinterpret_cast<const float4*>(w + i + 4);
  v8h o;
  o[0] = (_Float16)quant_e4m3(a.x); o[1] = (_Float16)quant_e4m3(a.y);
  o[2] = (_Float16)quant_e4m3(a.z); o[3] = (_Float16)quant_e4m3(a.w);
  o[4] = (_Float16)quant_e4m3(b.x); o[5] = (_Float16)quant_e4m3(b.y);
  o[6] = (_Float16)quant_e4m3(b.z); o[7] = (_Float16)quant_e4m3(b.w);
  *reinterpret_cast<v8h*>(w16 + i) = o;
}

// --- X: fp32 -> f16 (RNE), 8 elems/thread -----------------------------------
__global__ __launch_bounds__(256) void cvt_x_kernel(const float* __restrict__ x,
                                                    _Float16* __restrict__ x16) {
  size_t i = ((size_t)blockIdx.x * blockDim.x + threadIdx.x) * 8;
  float4 a = *reinterpret_cast<const float4*>(x + i);
  float4 b = *reinterpret_cast<const float4*>(x + i + 4);
  v8h o;
  o[0] = (_Float16)a.x; o[1] = (_Float16)a.y; o[2] = (_Float16)a.z; o[3] = (_Float16)a.w;
  o[4] = (_Float16)b.x; o[5] = (_Float16)b.y; o[6] = (_Float16)b.z; o[7] = (_Float16)b.w;
  *reinterpret_cast<v8h*>(x16 + i) = o;
}

// --- WMMA GEMM: C[M,N] = X16[M,K] * W16[N,K]^T + e4m3(bias) -----------------
__global__ __launch_bounds__(256, 1) void gemm_kernel(const _Float16* __restrict__ X16,
                                                      const _Float16* __restrict__ W16,
                                                      const float* __restrict__ bias,
                                                      float* __restrict__ out) {
#if USE_ASYNC_LDS
  __shared__ __align__(16) _Float16 As[2 * BM * LDA];   // 20 KB (double buffered)
  __shared__ __align__(16) _Float16 Bs[2 * BN * LDB];   // 40 KB
#else
  __shared__ __align__(16) _Float16 As[BM * LDA];
  __shared__ __align__(16) _Float16 Bs[BN * LDB];
#endif

  const int tid   = threadIdx.x;
  const int lane  = tid & 31;
  const int wave  = tid >> 5;        // 0..7
  const int lo    = lane & 15;       // N/M index within fragment
  const int hi    = lane >> 4;       // K half-selector per ISA layouts

  const int waveM = wave & 1;        // 0..1  -> 64-row slab
  const int waveN = wave >> 1;       // 0..3  -> 64-col slab

  const int blockM = blockIdx.y * BM;
  const int blockN = blockIdx.x * BN;

  v8f zero = {};
  v8f acc[4][4];
#pragma unroll
  for (int i = 0; i < 4; ++i)
#pragma unroll
    for (int j = 0; j < 4; ++j) acc[i][j] = zero;

  // 16 WMMAs on one K-step from LDS tile bases Ab/Bb.
  auto mmaStep = [&](const _Float16* Ab, const _Float16* Bb) {
    union V16 { v16h v; v8h h[2]; };
    V16 a[4], b[4];
    // A 16x32 f16 fragment: lanes 0-15 hold K 0-7 & 16-23, lanes 16-31 hold 8-15 & 24-31
#pragma unroll
    for (int i = 0; i < 4; ++i) {
      int r = waveM * 64 + i * 16 + lo;
      a[i].h[0] = *reinterpret_cast<const v8h*>(&Ab[r * LDA + hi * 8]);
      a[i].h[1] = *reinterpret_cast<const v8h*>(&Ab[r * LDA + hi * 8 + 16]);
    }
    // B 32x16 f16 fragment: lane = column n (= W row), lanes 0-15 K 0-7 & 16-23, 16-31 K 8-15 & 24-31
#pragma unroll
    for (int j = 0; j < 4; ++j) {
      int r = waveN * 64 + j * 16 + lo;
      b[j].h[0] = *reinterpret_cast<const v8h*>(&Bb[r * LDB + hi * 8]);
      b[j].h[1] = *reinterpret_cast<const v8h*>(&Bb[r * LDB + hi * 8 + 16]);
    }
#pragma unroll
    for (int i = 0; i < 4; ++i)
#pragma unroll
      for (int j = 0; j < 4; ++j)
        acc[i][j] = __builtin_amdgcn_wmma_f32_16x16x32_f16(
            false, a[i].v, false, b[j].v, (short)0, acc[i][j], false, false);
  };

#if USE_ASYNC_LDS
  // ----- async global->LDS DMA path (ASYNCcnt), one barrier per K-step -----
  auto issueTile = [&](int k0, int buf) {
    _Float16* Ab = &As[buf * BM * LDA];
    _Float16* Bb = &Bs[buf * BN * LDB];
#pragma unroll
    for (int s = 0; s < 2; ++s) {                     // A: 512 x 16B segs / 256 thr
      int seg = tid + s * 256;
      int r = seg >> 2, c = (seg & 3) * 8;
      __builtin_amdgcn_global_load_async_to_lds_b128(
          AS1_I4(X16 + (size_t)(blockM + r) * Kdim + k0 + c),
          AS3_I4(&Ab[r * LDA + c]), 0, 0);
    }
#pragma unroll
    for (int s = 0; s < 4; ++s) {                     // B: 1024 x 16B segs / 256 thr
      int seg = tid + s * 256;
      int r = seg >> 2, c = (seg & 3) * 8;
      __builtin_amdgcn_global_load_async_to_lds_b128(
          AS1_I4(W16 + (size_t)(blockN + r) * Kdim + k0 + c),
          AS3_I4(&Bb[r * LDB + c]), 0, 0);
    }
  };

  int buf = 0;
  issueTile(0, 0);
  for (int k0 = 0; k0 < Kdim; k0 += BK) {
    __builtin_amdgcn_s_wait_asynccnt(0);   // my DMA into As/Bs[buf] landed
    __syncthreads();                        // everyone's landed; prev reads done
    if (k0 + BK < Kdim) issueTile(k0 + BK, buf ^ 1);  // DMA overlaps the WMMAs
    mmaStep(&As[buf * BM * LDA], &Bs[buf * BN * LDB]);
    buf ^= 1;
  }
#else
  // ----- fallback: register-staged global->LDS, two barriers per K-step ----
  v8h aReg[2], bReg[4];
  auto loadGlobal = [&](int k0) {
#pragma unroll
    for (int s = 0; s < 2; ++s) {
      int seg = tid + s * 256;
      int r = seg >> 2, c = (seg & 3) * 8;
      aReg[s] = *reinterpret_cast<const v8h*>(X16 + (size_t)(blockM + r) * Kdim + k0 + c);
    }
#pragma unroll
    for (int s = 0; s < 4; ++s) {
      int seg = tid + s * 256;
      int r = seg >> 2, c = (seg & 3) * 8;
      bReg[s] = *reinterpret_cast<const v8h*>(W16 + (size_t)(blockN + r) * Kdim + k0 + c);
    }
  };
  auto stashLDS = [&]() {
#pragma unroll
    for (int s = 0; s < 2; ++s) {
      int seg = tid + s * 256;
      int r = seg >> 2, c = (seg & 3) * 8;
      *reinterpret_cast<v8h*>(&As[r * LDA + c]) = aReg[s];
    }
#pragma unroll
    for (int s = 0; s < 4; ++s) {
      int seg = tid + s * 256;
      int r = seg >> 2, c = (seg & 3) * 8;
      *reinterpret_cast<v8h*>(&Bs[r * LDB + c]) = bReg[s];
    }
  };

  loadGlobal(0);
  for (int k0 = 0; k0 < Kdim; k0 += BK) {
    __syncthreads();
    stashLDS();
    __syncthreads();
    if (k0 + BK < Kdim) loadGlobal(k0 + BK);
    mmaStep(As, Bs);
  }
#endif

  // Epilogue: add e4m3-quantized bias, scatter per C/D layout (VGPR g -> row g / g+8)
  float bq[4];
#pragma unroll
  for (int j = 0; j < 4; ++j)
    bq[j] = quant_e4m3(bias[blockN + waveN * 64 + j * 16 + lo]);

#pragma unroll
  for (int i = 0; i < 4; ++i) {
#pragma unroll
    for (int j = 0; j < 4; ++j) {
      int n = blockN + waveN * 64 + j * 16 + lo;
#pragma unroll
      for (int g = 0; g < 8; ++g) {
        int m = blockM + waveM * 64 + i * 16 + hi * 8 + g;
        out[(size_t)m * Ndim + n] = acc[i][j][g] + bq[j];
      }
    }
  }
}

extern "C" void kernel_launch(void* const* d_in, const int* in_sizes, int n_in,
                              void* d_out, int out_size, void* d_ws, size_t ws_size,
                              hipStream_t stream) {
  (void)in_sizes; (void)n_in; (void)out_size; (void)ws_size;
  const float* x    = (const float*)d_in[0];   // [2,4096,4096] fp32
  const float* w    = (const float*)d_in[1];   // [4096,4096]   fp32
  const float* bias = (const float*)d_in[2];   // [4096]        fp32
  float* out        = (float*)d_out;           // [2,4096,4096] fp32

  _Float16* X16 = (_Float16*)d_ws;                                    // 64 MB
  _Float16* W16 = (_Float16*)((char*)d_ws + (size_t)Mdim * Kdim * 2); // +32 MB

  // 1) x fp32 -> f16          (M*K/8 threads)
  cvt_x_kernel<<<(Mdim * (size_t)Kdim) / 8 / 256, 256, 0, stream>>>(x, X16);
  // 2) W fp32 -> e4m3 -> f16  (N*K/8 threads)
  quant_w_kernel<<<((size_t)Ndim * Kdim) / 8 / 256, 256, 0, stream>>>(w, W16);
  // 3) WMMA GEMM
  dim3 grid(Ndim / BN, Mdim / BM);   // (16, 64)
  gemm_kernel<<<grid, 256, 0, stream>>>(X16, W16, bias, out);
}
